// GraphMamba_11699490914481
// MI455X (gfx1250) — matrix operand order repaired and checked by reading, not compiled, not run on hardware
//
#include <hip/hip_runtime.h>
#include <hip/hip_bf16.h>

// ---------------- types ----------------
typedef __bf16 bf16_t;
typedef __attribute__((ext_vector_type(16))) bf16_t v16bf;
typedef __attribute__((ext_vector_type(8)))  float  v8f;
typedef __attribute__((ext_vector_type(4)))  int    v4i;
typedef __attribute__((address_space(1))) v4i* gptr_v4i;   // global int4 vector ptr
typedef __attribute__((address_space(3))) v4i* lptr_v4i;   // LDS int4 vector ptr

#define HID   128
#define NODES 100000
#define EDGES 640000
#define MTILES (NODES / 16)   // 6250 exact

union Frag {
  v16bf        bf;
  unsigned int u32[8];
  uint4        q[2];
};

__device__ __forceinline__ unsigned short f32_to_bf16(float f) {
  unsigned int u = __float_as_uint(f);
  unsigned int r = 0x7FFFu + ((u >> 16) & 1u);   // round-to-nearest-even
  u += r;
  return (unsigned short)(u >> 16);
}

// ---------------- kernel 1: zero agg + deg ----------------
__global__ void gm_zero_kernel(float* __restrict__ agg, float* __restrict__ deg) {
  const int stride = gridDim.x * blockDim.x;
  for (int i = blockIdx.x * blockDim.x + threadIdx.x; i < NODES * HID; i += stride)
    agg[i] = 0.0f;
  for (int i = blockIdx.x * blockDim.x + threadIdx.x; i < NODES; i += stride)
    deg[i] = 0.0f;
}

// ---------------- kernel 2: weights fp32 -> bf16 ----------------
__global__ void gm_cvt_weights_kernel(const float* __restrict__ Win,
                                      const float* __restrict__ Wg,
                                      const float* __restrict__ Wo,
                                      unsigned short* __restrict__ wb) {
  const int i = blockIdx.x * blockDim.x + threadIdx.x;
  if (i < HID * HID) {
    wb[i]                 = f32_to_bf16(Win[i]);
    wb[HID * HID + i]     = f32_to_bf16(Wg[i]);
    wb[2 * HID * HID + i] = f32_to_bf16(Wo[i]);
  }
}

// ---------------- kernel 3: edge gather + scatter-add ----------------
// one wave32 per edge: lane loads float4 of X[src], atomic-adds into agg[dst]
__global__ void gm_scatter_kernel(const float* __restrict__ X,
                                  const long long* __restrict__ edges, // [2][E] int64
                                  float* __restrict__ agg,
                                  float* __restrict__ deg) {
  const int gid  = blockIdx.x * blockDim.x + threadIdx.x;
  const int edge = gid >> 5;
  const int lane = gid & 31;
  if (edge >= EDGES) return;
  const long long s = edges[edge];
  const long long d = edges[EDGES + edge];
  const float4 v = *(const float4*)(X + (size_t)s * HID + lane * 4);
  float* ap = agg + (size_t)d * HID + lane * 4;
  atomicAdd(ap + 0, v.x);
  atomicAdd(ap + 1, v.y);
  atomicAdd(ap + 2, v.z);
  atomicAdd(ap + 3, v.w);
  if (lane == 0) atomicAdd(deg + d, 1.0f);
}

// ---------------- kernel 4: normalize by clamped degree, -> bf16 ----------------
__global__ void gm_normalize_kernel(const float* __restrict__ agg,
                                    const float* __restrict__ deg,
                                    unsigned short* __restrict__ aggb) {
  const int gid  = blockIdx.x * blockDim.x + threadIdx.x;
  const int node = gid >> 5;
  const int lane = gid & 31;
  if (node >= NODES) return;
  float dg  = deg[node];
  float inv = 1.0f / (dg < 1.0f ? 1.0f : dg);
  const float4 v = *(const float4*)(agg + (size_t)node * HID + lane * 4);
  ushort4 o;
  o.x = f32_to_bf16(v.x * inv);
  o.y = f32_to_bf16(v.y * inv);
  o.z = f32_to_bf16(v.z * inv);
  o.w = f32_to_bf16(v.w * inv);
  *(ushort4*)(aggb + (size_t)node * HID + lane * 4) = o;
}

// ---------------- kernel 5: fused gated-MLP via WMMA bf16 ----------------
// one wave per 16-row node tile; three GEMMs fused with LDS staging of both
// the input tile (async global->LDS copy) and the relu(h)*sigmoid(g) activation.
__global__ __launch_bounds__(256) void gm_mlp_wmma_kernel(
    const unsigned short* __restrict__ aggb,   // [NODES][HID] bf16
    const unsigned short* __restrict__ wb,     // [3][HID][HID] bf16 (Win,Wg,Wo row-major)
    const float* __restrict__ b_in,
    const float* __restrict__ b_gate,
    const float* __restrict__ b_out,
    float* __restrict__ out) {                 // [NODES][HID] f32
  __shared__ unsigned short act[8][16 * HID];  // per-wave 16x128 bf16 tile (input, then activation)

  const int wave = threadIdx.x >> 5;
  const int lane = threadIdx.x & 31;
  const int half = lane >> 4;   // 0: lanes 0-15, 1: lanes 16-31
  const int l16  = lane & 15;

  const int mtile = blockIdx.x * 8 + wave;
  if (mtile >= MTILES) return;   // wave-uniform exit, EXEC stays all-ones for WMMA
  const int m0 = mtile * 16;

  const unsigned short* Win = wb;
  const unsigned short* Wg  = wb + HID * HID;
  const unsigned short* Wo  = wb + 2 * HID * HID;

  // --- coalesced per-wave copy of the 16x128 bf16 agg tile into LDS ---
  // 4 KB per wave = 256 x b128; each lane moves 8 x 16B via async global->LDS.
  {
    const size_t gbase = (size_t)m0 * HID;  // ushort index of tile start
#if __has_builtin(__builtin_amdgcn_global_load_async_to_lds_b128)
#pragma unroll
    for (int j = 0; j < 8; ++j) {
      const int i = lane + 32 * j;          // b128 index within tile
      __builtin_amdgcn_global_load_async_to_lds_b128(
          (gptr_v4i)(aggb + gbase + (size_t)i * 8),
          (lptr_v4i)&act[wave][i * 8],
          0, 0);
    }
#if __has_builtin(__builtin_amdgcn_s_wait_asynccnt)
    __builtin_amdgcn_s_wait_asynccnt(0);
#else
    asm volatile("s_wait_asynccnt 0" ::: "memory");
#endif
#else
#pragma unroll
    for (int j = 0; j < 8; ++j) {
      const int i = lane + 32 * j;
      *(uint4*)&act[wave][i * 8] = *(const uint4*)(aggb + gbase + (size_t)i * 8);
    }
#endif
  }

  // --- A fragments of agg tile (16x32 bf16 layout) from LDS, 4 K-chunks of 32 ---
  Frag afr[4];
#pragma unroll
  for (int kc = 0; kc < 4; ++kc) {
    const int k0 = kc * 32;
#pragma unroll
    for (int v = 0; v < 8; ++v) {
      const int k = k0 + half * 16 + v * 2;
      afr[kc].u32[v] = *(const unsigned int*)&act[wave][l16 * HID + k];
    }
  }

  // --- stage A: h = A*Win^T + b_in ; g = A*Wg^T + b_gate ; act = relu(h)*sigmoid(g) ---
#pragma unroll
  for (int nt = 0; nt < 8; ++nt) {
    const int n0 = nt * 16;
    v8f ah = {};
    v8f ag = {};
#pragma unroll
    for (int kc = 0; kc < 4; ++kc) {
      const int k0 = kc * 32;
      // B[k][n] = W[n][k]: lane l16 -> column n0+l16, 16 contiguous k -> 2x b128 loads
      Frag bh, bg;
      const uint4* ph = (const uint4*)(Win + (size_t)(n0 + l16) * HID + k0 + half * 16);
      bh.q[0] = ph[0]; bh.q[1] = ph[1];
      const uint4* pg = (const uint4*)(Wg + (size_t)(n0 + l16) * HID + k0 + half * 16);
      bg.q[0] = pg[0]; bg.q[1] = pg[1];
      ah = __builtin_amdgcn_wmma_f32_16x16x32_bf16(false, afr[kc].bf, false, bh.bf,
                                                   (short)0, ah, false, false);
      ag = __builtin_amdgcn_wmma_f32_16x16x32_bf16(false, afr[kc].bf, false, bg.bf,
                                                   (short)0, ag, false, false);
    }
    const float bi  = b_in[n0 + l16];
    const float bg2 = b_gate[n0 + l16];
#pragma unroll
    for (int r = 0; r < 8; ++r) {
      float h = ah[r] + bi;
      float g = ag[r] + bg2;
      h = h > 0.0f ? h : 0.0f;
      const float s = 1.0f / (1.0f + __expf(-g));
      // D layout: M = r + 8*half, N = l16 (per 32-bit 16x16 C/D layout)
      act[wave][(r + 8 * half) * HID + n0 + l16] = f32_to_bf16(h * s);
    }
  }

  // --- stage B: out = act * Wo^T + b_out ; A fragments re-read from LDS ---
  Frag cfr[4];
#pragma unroll
  for (int kc = 0; kc < 4; ++kc) {
    const int k0 = kc * 32;
#pragma unroll
    for (int v = 0; v < 8; ++v) {
      const int k = k0 + half * 16 + v * 2;
      cfr[kc].u32[v] = *(const unsigned int*)&act[wave][l16 * HID + k];
    }
  }

#pragma unroll
  for (int nt = 0; nt < 8; ++nt) {
    const int n0 = nt * 16;
    v8f ao = {};
#pragma unroll
    for (int kc = 0; kc < 4; ++kc) {
      const int k0 = kc * 32;
      Frag bo;
      const uint4* po = (const uint4*)(Wo + (size_t)(n0 + l16) * HID + k0 + half * 16);
      bo.q[0] = po[0]; bo.q[1] = po[1];
      ao = __builtin_amdgcn_wmma_f32_16x16x32_bf16(false, cfr[kc].bf, false, bo.bf,
                                                   (short)0, ao, false, false);
    }
    const float bov = b_out[n0 + l16];
#pragma unroll
    for (int r = 0; r < 8; ++r) {
      out[(size_t)(m0 + r + 8 * half) * HID + n0 + l16] = ao[r] + bov;
    }
  }
}

// ---------------- launcher ----------------
extern "C" void kernel_launch(void* const* d_in, const int* in_sizes, int n_in,
                              void* d_out, int out_size, void* d_ws, size_t ws_size,
                              hipStream_t stream) {
  (void)in_sizes; (void)n_in; (void)out_size; (void)ws_size;

  const float*     X      = (const float*)d_in[0];
  const long long* edges  = (const long long*)d_in[1];   // int64 [2][E]
  const float*     W_in   = (const float*)d_in[2];
  const float*     b_in   = (const float*)d_in[3];
  const float*     W_gate = (const float*)d_in[4];
  const float*     b_gate = (const float*)d_in[5];
  const float*     W_out  = (const float*)d_in[6];
  const float*     b_out  = (const float*)d_in[7];
  float*           out    = (float*)d_out;

  // workspace layout (256B aligned)
  char* ws = (char*)d_ws;
  size_t off = 0;
  auto take = [&](size_t bytes) -> char* {
    char* p = ws + off;
    off = (off + bytes + 255) & ~(size_t)255;
    return p;
  };
  float*          agg  = (float*)take((size_t)NODES * HID * sizeof(float));
  float*          deg  = (float*)take((size_t)NODES * sizeof(float));
  unsigned short* aggb = (unsigned short*)take((size_t)NODES * HID * sizeof(unsigned short));
  unsigned short* wbuf = (unsigned short*)take((size_t)3 * HID * HID * sizeof(unsigned short));

  // 1) zero accumulators
  gm_zero_kernel<<<2048, 256, 0, stream>>>(agg, deg);

  // 2) weights -> bf16
  gm_cvt_weights_kernel<<<(HID * HID + 255) / 256, 256, 0, stream>>>(W_in, W_gate, W_out, wbuf);

  // 3) edge gather + atomic scatter-add (one wave per edge)
  gm_scatter_kernel<<<(EDGES * 32) / 256, 256, 0, stream>>>(X, edges, agg, deg);

  // 4) mean-normalize + bf16 convert
  gm_normalize_kernel<<<(NODES * 32 + 255) / 256, 256, 0, stream>>>(agg, deg, aggb);

  // 5) fused gated-MLP (WMMA bf16, f32 accumulate)
  gm_mlp_wmma_kernel<<<(MTILES + 7) / 8, 256, 0, stream>>>(aggb, wbuf, b_in, b_gate, b_out, out);
}
